// MOETransformerBlock_69664369541805
// MI455X (gfx1250) — matrix operand order repaired and checked by reading, not compiled, not run on hardware
//
#include <hip/hip_runtime.h>
#include <hip/hip_bf16.h>
#include <stdint.h>

// ---------------------------------------------------------------------------
// MI455X (gfx1250) transformer block: LN -> QKV -> flash attention -> O-proj
// -> LN -> static-split MoE MLP.  All matmuls on v_wmma_f32_16x16x32_bf16.
// GEMM uses double-buffered LDS fed by gfx1250 async global->LDS loads.
// ---------------------------------------------------------------------------

#define USE_ASYNC_LDS 1   // set 0 to fall back to synchronous LDS staging

typedef __bf16 bf16;
typedef __attribute__((ext_vector_type(16))) __bf16 v16bf;
typedef __attribute__((ext_vector_type(8)))  float  v8f;

static __device__ inline v8f wmma_bf16(v16bf a, v16bf b, v8f c) {
  // 8 args: (neg_a, A, neg_b, B, c_mod, C, reuse_a, reuse_b)
  return __builtin_amdgcn_wmma_f32_16x16x32_bf16(false, a, false, b, (short)0, c,
                                                 false, false);
}

static constexpr int kS     = 2048;
static constexpr int kChunk = 256;   // tokens per (batch, expert)

// ---------------------------------------------------------------------------
// fp32 -> bf16 conversion (weights, once per call)
// ---------------------------------------------------------------------------
__global__ void __launch_bounds__(256)
cvt_f32_bf16(const float* __restrict__ in, bf16* __restrict__ out, size_t n) {
  size_t i  = (size_t)blockIdx.x * blockDim.x + threadIdx.x;
  size_t st = (size_t)gridDim.x * blockDim.x;
  for (; i < n; i += st) out[i] = (bf16)in[i];
}

// ---------------------------------------------------------------------------
// Row LayerNorm, fp32 in -> bf16 out (one 256-thread block per row)
// ---------------------------------------------------------------------------
__global__ void __launch_bounds__(256)
layernorm_bf16(const float* __restrict__ x, const float* __restrict__ g,
               const float* __restrict__ b, bf16* __restrict__ out, int D) {
  __shared__ float red[256];
  const int tid = threadIdx.x;
  const float* xr = x + (size_t)blockIdx.x * D;

  float s = 0.f;
  for (int i = tid; i < D; i += 256) s += xr[i];
  red[tid] = s; __syncthreads();
  for (int st = 128; st > 0; st >>= 1) {
    if (tid < st) red[tid] += red[tid + st];
    __syncthreads();
  }
  const float mean = red[0] / (float)D;
  __syncthreads();

  float vs = 0.f;
  for (int i = tid; i < D; i += 256) { float d = xr[i] - mean; vs += d * d; }
  red[tid] = vs; __syncthreads();
  for (int st = 128; st > 0; st >>= 1) {
    if (tid < st) red[tid] += red[tid + st];
    __syncthreads();
  }
  const float rstd = rsqrtf(red[0] / (float)D + 1e-5f);

  bf16* orow = out + (size_t)blockIdx.x * D;
  for (int i = tid; i < D; i += 256)
    orow[i] = (bf16)((xr[i] - mean) * rstd * g[i] + b[i]);
}

// ---------------------------------------------------------------------------
// bf16 GEMM: C[M,N] = epilogue(A[M,K] @ B[K,N] + bias [+ resid])
// 128x128x32 block tiles, 8 waves, each wave -> 32x64 strip = 8 WMMA tiles.
// Double-buffered LDS; fills via GLOBAL_LOAD_ASYNC_TO_LDS_B128 (ASYNCcnt).
// MOE mode: gridDim.z enumerates 16 static (batch, expert) token chunks.
// ---------------------------------------------------------------------------
template <bool SILU, bool RES, bool OUT_BF, bool MOE>
__global__ void __launch_bounds__(256)
gemm_bf16_wmma(const bf16* __restrict__ A, const bf16* __restrict__ Bm,
               const float* __restrict__ bias, const float* __restrict__ resid,
               float* __restrict__ Cf, bf16* __restrict__ Cb,
               int M, int N, int K) {
  (void)M;
  constexpr int BM = 128, BN = 128, BK = 32;
  constexpr int LDA = BK + 16;   //  96 B row stride (16B aligned)
  constexpr int LDB = BN + 16;   // 288 B row stride (16B aligned)
  __shared__ bf16 As[2][BM][LDA];
  __shared__ bf16 Bs[2][BK][LDB];

  if constexpr (MOE) {
    const int ez = blockIdx.z >> 1;          // expert
    const int bz = blockIdx.z & 1;           // batch
    const size_t row0 = (size_t)bz * kS + (size_t)ez * kChunk;
    A    += row0 * (size_t)K;
    Bm   += (size_t)ez * (size_t)K * N;
    bias += (size_t)ez * N;
    if constexpr (RES)    resid += row0 * (size_t)N;
    if constexpr (OUT_BF) Cb    += row0 * (size_t)N;
    else                  Cf    += row0 * (size_t)N;
  }

  const int tid  = threadIdx.x;
  const int lane = tid & 31;
  const int wv   = tid >> 5;
  const int half = lane >> 4;
  const int ln   = lane & 15;
  const int wr   = wv & 3;              // 4 row strips of 32
  const int wc   = wv >> 2;             // 2 col strips of 64
  const int m0   = blockIdx.y * BM;
  const int n0   = blockIdx.x * BN;

  const int arow = tid >> 2, acol = (tid & 3) * 8;    // A: 128x32, +64-row pair
  const int brow = tid >> 4, bcol = (tid & 15) * 8;   // B:  32x128, +16-row pair

  v8f acc[2][4] = {};

  auto issue_tile = [&](int kt, int buf) {
#if USE_ASYNC_LDS
    const uint32_t la0 = (uint32_t)(uintptr_t)&As[buf][arow][acol];
    const uint32_t la1 = (uint32_t)(uintptr_t)&As[buf][arow + 64][acol];
    const uint32_t lb0 = (uint32_t)(uintptr_t)&Bs[buf][brow][bcol];
    const uint32_t lb1 = (uint32_t)(uintptr_t)&Bs[buf][brow + 16][bcol];
    const uint32_t ga0 = (uint32_t)(((size_t)(m0 + arow)      * K + kt + acol) * sizeof(bf16));
    const uint32_t ga1 = (uint32_t)(((size_t)(m0 + arow + 64) * K + kt + acol) * sizeof(bf16));
    const uint32_t gb0 = (uint32_t)(((size_t)(kt + brow)      * N + n0 + bcol) * sizeof(bf16));
    const uint32_t gb1 = (uint32_t)(((size_t)(kt + brow + 16) * N + n0 + bcol) * sizeof(bf16));
    asm volatile(
        "global_load_async_to_lds_b128 %0, %4, %8\n\t"
        "global_load_async_to_lds_b128 %1, %5, %8\n\t"
        "global_load_async_to_lds_b128 %2, %6, %9\n\t"
        "global_load_async_to_lds_b128 %3, %7, %9"
        :
        : "v"(la0), "v"(la1), "v"(lb0), "v"(lb1),
          "v"(ga0), "v"(ga1), "v"(gb0), "v"(gb1),
          "s"(A), "s"(Bm)
        : "memory");
#else
    *reinterpret_cast<uint4*>(&As[buf][arow][acol]) =
        *reinterpret_cast<const uint4*>(&A[(size_t)(m0 + arow) * K + kt + acol]);
    *reinterpret_cast<uint4*>(&As[buf][arow + 64][acol]) =
        *reinterpret_cast<const uint4*>(&A[(size_t)(m0 + arow + 64) * K + kt + acol]);
    *reinterpret_cast<uint4*>(&Bs[buf][brow][bcol]) =
        *reinterpret_cast<const uint4*>(&Bm[(size_t)(kt + brow) * N + n0 + bcol]);
    *reinterpret_cast<uint4*>(&Bs[buf][brow + 16][bcol]) =
        *reinterpret_cast<const uint4*>(&Bm[(size_t)(kt + brow + 16) * N + n0 + bcol]);
#endif
  };

  issue_tile(0, 0);

  const int nkt = K / BK;
  for (int it = 0; it < nkt; ++it) {
    const int cur = it & 1;
#if USE_ASYNC_LDS
    asm volatile("s_wait_asynccnt 0x0" ::: "memory");
#endif
    __syncthreads();
    if (it + 1 < nkt) issue_tile((it + 1) * BK, cur ^ 1);

    // A fragments (ISA 7.12.2, 16-bit A 16x32): lane m=ln, K split 8/8 by half
    v16bf afr[2];
#pragma unroll
    for (int t = 0; t < 2; ++t) {
#pragma unroll
      for (int j = 0; j < 16; ++j) {
        const int kk = j + 8 * half + ((j >= 8) ? 8 : 0);
        afr[t][j] = As[cur][wr * 32 + t * 16 + ln][kk];
      }
    }
    // B fragments: element j -> K row j + 8*half + (j>=8 ? 8 : 0), column n = ln
#pragma unroll
    for (int u = 0; u < 4; ++u) {
      v16bf bfr;
#pragma unroll
      for (int j = 0; j < 16; ++j)
        bfr[j] = Bs[cur][j + 8 * half + ((j >= 8) ? 8 : 0)][wc * 64 + u * 16 + ln];
#pragma unroll
      for (int t = 0; t < 2; ++t)
        acc[t][u] = wmma_bf16(afr[t], bfr, acc[t][u]);
    }
  }

  // Epilogue. C/D layout: reg r -> row r + 8*half, col = lane n.
  const int orow0 = m0 + wr * 32 + 8 * half;
#pragma unroll
  for (int t = 0; t < 2; ++t) {
#pragma unroll
    for (int u = 0; u < 4; ++u) {
      const int col = n0 + wc * 64 + u * 16 + ln;
      const float bb = bias[col];
#pragma unroll
      for (int r = 0; r < 8; ++r) {
        const int row = orow0 + t * 16 + r;
        float v = acc[t][u][r] + bb;
        if constexpr (SILU) v = v / (1.f + __expf(-v));   // silu = x*sigmoid(x)
        if constexpr (RES)  v += resid[(size_t)row * N + col];
        if constexpr (OUT_BF) Cb[(size_t)row * N + col] = (bf16)v;
        else                  Cf[(size_t)row * N + col] = v;
      }
    }
  }
}

// ---------------------------------------------------------------------------
// Flash attention, Dh = 64. One workgroup = 8 waves = 128 query rows of one
// (batch, head).  K/V staged through LDS in 32-key blocks; scores & P*V on
// WMMA; online softmax state (m, l) held in registers per wave.
// ---------------------------------------------------------------------------
__global__ void __launch_bounds__(256)
attn_flash_bf16(const bf16* __restrict__ q, const bf16* __restrict__ k,
                const bf16* __restrict__ v, bf16* __restrict__ o,
                int S, int D) {
  constexpr int DH = 64;
  __shared__ bf16 Ks[32][DH + 16];
  __shared__ bf16 Vs[32][DH + 16];
  __shared__ bf16 Pl[8][16][40];     // per-wave P re-layout buffer

  const int tid  = threadIdx.x;
  const int lane = tid & 31;
  const int wv   = tid >> 5;
  const int half = lane >> 4;
  const int ln   = lane & 15;
  const int qb   = blockIdx.x * 128;
  const int hd   = blockIdx.y;
  const int bt   = blockIdx.z;
  const size_t bs = (size_t)bt * S;

  // Preload this wave's Q fragments (16 rows x 64 dims -> 2 A-frags)
  const int qrow = qb + wv * 16 + ln;
  const bf16* qp = q + (bs + qrow) * D + hd * DH;
  v16bf qa0, qa1;
#pragma unroll
  for (int j = 0; j < 16; ++j) {
    const int kk = j + 8 * half + ((j >= 8) ? 8 : 0);
    qa0[j] = qp[kk];
    qa1[j] = qp[32 + kk];
  }

  float mstat[8], lstat[8];
  v8f oacc[4] = {};
#pragma unroll
  for (int r = 0; r < 8; ++r) { mstat[r] = -1e30f; lstat[r] = 0.f; }

  const int ckey = tid >> 3;          // 0..31
  const int cd   = (tid & 7) * 8;     // 0..56

  for (int kb = 0; kb < S; kb += 32) {
    *reinterpret_cast<uint4*>(&Ks[ckey][cd]) =
        *reinterpret_cast<const uint4*>(&k[(bs + kb + ckey) * D + hd * DH + cd]);
    *reinterpret_cast<uint4*>(&Vs[ckey][cd]) =
        *reinterpret_cast<const uint4*>(&v[(bs + kb + ckey) * D + hd * DH + cd]);
    __syncthreads();

    // scores S16x32 = Q(16x64) @ K^T : 2 k-steps x 2 key tiles
    v8f s0 = {}, s1 = {};
#pragma unroll
    for (int kit = 0; kit < 2; ++kit) {
      v16bf b0, b1;
#pragma unroll
      for (int j = 0; j < 16; ++j) {
        const int dd = kit * 32 + j + 8 * half + ((j >= 8) ? 8 : 0);
        b0[j] = Ks[ln][dd];        // key tile 0: keys 0..15
        b1[j] = Ks[16 + ln][dd];   // key tile 1: keys 16..31
      }
      const v16bf a = kit ? qa1 : qa0;
      s0 = wmma_bf16(a, b0, s0);
      s1 = wmma_bf16(a, b1, s1);
    }

    // online softmax over this 32-key block
#pragma unroll
    for (int r = 0; r < 8; ++r) {
      const float x0 = s0[r] * 0.125f;   // 1/sqrt(64)
      const float x1 = s1[r] * 0.125f;
      float mx = fmaxf(x0, x1);
#pragma unroll
      for (int d = 1; d < 16; d <<= 1) mx = fmaxf(mx, __shfl_xor(mx, d, 32));
      const float mnew  = fmaxf(mstat[r], mx);
      const float alpha = __expf(mstat[r] - mnew);
      const float p0 = __expf(x0 - mnew);
      const float p1 = __expf(x1 - mnew);
      float rs = p0 + p1;
#pragma unroll
      for (int d = 1; d < 16; d <<= 1) rs += __shfl_xor(rs, d, 32);
      lstat[r] = lstat[r] * alpha + rs;
      mstat[r] = mnew;
#pragma unroll
      for (int c = 0; c < 4; ++c) oacc[c][r] *= alpha;
      Pl[wv][8 * half + r][ln]      = (bf16)p0;
      Pl[wv][8 * half + r][16 + ln] = (bf16)p1;
    }
    __syncthreads();   // P visible; also fences before Ks/Vs reuse below

    // O += P(16x32) @ V(32x64)
    v16bf pa;
#pragma unroll
    for (int j = 0; j < 16; ++j) {
      const int kk = j + 8 * half + ((j >= 8) ? 8 : 0);
      pa[j] = Pl[wv][ln][kk];
    }
#pragma unroll
    for (int c = 0; c < 4; ++c) {
      v16bf bv;
#pragma unroll
      for (int j = 0; j < 16; ++j) bv[j] = Vs[j + 8 * half + ((j >= 8) ? 8 : 0)][c * 16 + ln];
      oacc[c] = wmma_bf16(pa, bv, oacc[c]);
    }
    __syncthreads();
  }

  // normalize and store (bf16) into O buffer
  bf16* op = o + (bs + qb + wv * 16) * D + hd * DH;
#pragma unroll
  for (int c = 0; c < 4; ++c) {
#pragma unroll
    for (int r = 0; r < 8; ++r) {
      const int row = 8 * half + r;
      op[(size_t)row * D + c * 16 + ln] = (bf16)(oacc[c][r] / lstat[r]);
    }
  }
}

// ---------------------------------------------------------------------------
// Host orchestration
// ---------------------------------------------------------------------------
extern "C" void kernel_launch(void* const* d_in, const int* in_sizes, int n_in,
                              void* d_out, int out_size, void* d_ws, size_t ws_size,
                              hipStream_t stream) {
  (void)in_sizes; (void)n_in; (void)out_size; (void)ws_size;
  constexpr int Bb = 2, S = kS, DIM = 1024, HEADS = 16, EXPERTS = 8;
  constexpr int DFF = 4 * DIM;
  constexpr int NTOK = Bb * S;                 // 4096

  const float* x     = (const float*)d_in[0];
  const float* ln1_g = (const float*)d_in[1];
  const float* ln1_b = (const float*)d_in[2];
  const float* ln2_g = (const float*)d_in[3];
  const float* ln2_b = (const float*)d_in[4];
  const float* wq = (const float*)d_in[5];  const float* bq = (const float*)d_in[6];
  const float* wk = (const float*)d_in[7];  const float* bk = (const float*)d_in[8];
  const float* wv = (const float*)d_in[9];  const float* bv = (const float*)d_in[10];
  const float* wo = (const float*)d_in[11]; const float* bo = (const float*)d_in[12];
  const float* w1 = (const float*)d_in[13]; const float* b1 = (const float*)d_in[14];
  const float* w2 = (const float*)d_in[15]; const float* b2 = (const float*)d_in[16];
  float* out = (float*)d_out;

  // workspace carve-up (256B aligned)
  uint8_t* ws = (uint8_t*)d_ws;
  size_t off = 0;
  auto carve = [&](size_t bytes) -> void* {
    void* p = ws + off;
    off += (bytes + 255) & ~(size_t)255;
    return p;
  };
  bf16* hbf  = (bf16*)carve((size_t)NTOK * DIM * 2);
  bf16* qbf  = (bf16*)carve((size_t)NTOK * DIM * 2);
  bf16* kbf  = (bf16*)carve((size_t)NTOK * DIM * 2);
  bf16* vbf  = (bf16*)carve((size_t)NTOK * DIM * 2);
  bf16* obf  = (bf16*)carve((size_t)NTOK * DIM * 2);
  float* x2  = (float*)carve((size_t)NTOK * DIM * 4);
  bf16* h2bf = (bf16*)carve((size_t)NTOK * DIM * 2);
  bf16* ffbf = (bf16*)carve((size_t)NTOK * DFF * 2);
  bf16* wqbf = (bf16*)carve((size_t)DIM * DIM * 2);
  bf16* wkbf = (bf16*)carve((size_t)DIM * DIM * 2);
  bf16* wvbf = (bf16*)carve((size_t)DIM * DIM * 2);
  bf16* wobf = (bf16*)carve((size_t)DIM * DIM * 2);
  bf16* w1bf = (bf16*)carve((size_t)EXPERTS * DIM * DFF * 2);
  bf16* w2bf = (bf16*)carve((size_t)EXPERTS * DFF * DIM * 2);

  // 1) weight conversions fp32 -> bf16
  {
    const size_t nw = (size_t)DIM * DIM;
    cvt_f32_bf16<<<1024, 256, 0, stream>>>(wq, wqbf, nw);
    cvt_f32_bf16<<<1024, 256, 0, stream>>>(wk, wkbf, nw);
    cvt_f32_bf16<<<1024, 256, 0, stream>>>(wv, wvbf, nw);
    cvt_f32_bf16<<<1024, 256, 0, stream>>>(wo, wobf, nw);
    const size_t ne = (size_t)EXPERTS * DIM * DFF;
    cvt_f32_bf16<<<4096, 256, 0, stream>>>(w1, w1bf, ne);
    cvt_f32_bf16<<<4096, 256, 0, stream>>>(w2, w2bf, ne);
  }

  const dim3 blk(256);

  // 2) LN1 -> hbf
  layernorm_bf16<<<NTOK, blk, 0, stream>>>(x, ln1_g, ln1_b, hbf, DIM);

  // 3) Q/K/V projections (bf16 out, +bias)
  {
    dim3 grid(DIM / 128, NTOK / 128);
    gemm_bf16_wmma<false, false, true, false><<<grid, blk, 0, stream>>>(
        hbf, wqbf, bq, nullptr, nullptr, qbf, NTOK, DIM, DIM);
    gemm_bf16_wmma<false, false, true, false><<<grid, blk, 0, stream>>>(
        hbf, wkbf, bk, nullptr, nullptr, kbf, NTOK, DIM, DIM);
    gemm_bf16_wmma<false, false, true, false><<<grid, blk, 0, stream>>>(
        hbf, wvbf, bv, nullptr, nullptr, vbf, NTOK, DIM, DIM);
  }

  // 4) flash attention -> obf
  {
    dim3 grid(S / 128, HEADS, Bb);
    attn_flash_bf16<<<grid, blk, 0, stream>>>(qbf, kbf, vbf, obf, S, DIM);
  }

  // 5) out-proj + residual(x) -> x2 (fp32)
  {
    dim3 grid(DIM / 128, NTOK / 128);
    gemm_bf16_wmma<false, true, false, false><<<grid, blk, 0, stream>>>(
        obf, wobf, bo, x, x2, nullptr, NTOK, DIM, DIM);
  }

  // 6) LN2 -> h2bf
  layernorm_bf16<<<NTOK, blk, 0, stream>>>(x2, ln2_g, ln2_b, h2bf, DIM);

  // 7) MoE, one launch per projection; gridDim.z = 16 (batch, expert) chunks
  {
    dim3 gridUp(DFF / 128, kChunk / 128, Bb * EXPERTS);
    gemm_bf16_wmma<true, false, true, true><<<gridUp, blk, 0, stream>>>(
        h2bf, w1bf, b1, nullptr, nullptr, ffbf, kChunk, DFF, DIM);

    dim3 gridDn(DIM / 128, kChunk / 128, Bb * EXPERTS);
    gemm_bf16_wmma<false, true, false, true><<<gridDn, blk, 0, stream>>>(
        ffbf, w2bf, b2, x2, out, nullptr, kChunk, DIM, DFF);
  }
}
